// AsymQuantMatMul_42520176230456
// MI455X (gfx1250) — 8-bit matrix operand order repaired and checked by reading, compiled, not run on hardware
//
#include <hip/hip_runtime.h>
#include <hip/hip_bf16.h>

typedef __attribute__((ext_vector_type(8))) int v8i;

__device__ __forceinline__ unsigned fenc(float f) {
    unsigned u = __float_as_uint(f);
    return (u & 0x80000000u) ? ~u : (u | 0x80000000u);
}
__device__ __forceinline__ float fdec(unsigned k) {
    unsigned u = (k & 0x80000000u) ? (k ^ 0x80000000u) : ~k;
    return __uint_as_float(u);
}
__device__ __forceinline__ unsigned umin2(unsigned a, unsigned b) { return a < b ? a : b; }
__device__ __forceinline__ unsigned umax2(unsigned a, unsigned b) { return a > b ? a : b; }

__global__ void init_keys_kernel(unsigned* keys) {
    keys[0] = 0xFFFFFFFFu;
    keys[1] = 0x00000000u;
    keys[2] = 0xFFFFFFFFu;
    keys[3] = 0x00000000u;
}

__global__ __launch_bounds__(256) void minmax_kernel(const float* __restrict__ x, int n4,
                                                     unsigned* __restrict__ keys) {
    __shared__ unsigned smn[256];
    __shared__ unsigned smx[256];
    const int t = threadIdx.x;
    float lmin = 3.402823466e38f, lmax = -3.402823466e38f;
    const float4* v = (const float4*)x;
    for (int i = blockIdx.x * 256 + t; i < n4; i += gridDim.x * 256) {
        float4 f = v[i];
        lmin = fminf(lmin, fminf(fminf(f.x, f.y), fminf(f.z, f.w)));
        lmax = fmaxf(lmax, fmaxf(fmaxf(f.x, f.y), fmaxf(f.z, f.w)));
    }
    smn[t] = fenc(lmin);
    smx[t] = fenc(lmax);
    __syncthreads();
    for (int off = 128; off > 0; off >>= 1) {
        if (t < off) {
            smn[t] = umin2(smn[t], smn[t + off]);
            smx[t] = umax2(smx[t], smx[t + off]);
        }
        __syncthreads();
    }
    if (t == 0) {
        atomicMin(&keys[0], smn[0]);
        atomicMax(&keys[1], smx[0]);
    }
}

__global__ void finalize_kernel(const unsigned* __restrict__ keys, float* __restrict__ scal) {
    float minA = fdec(keys[0]), maxA = fdec(keys[1]);
    float minB = fdec(keys[2]), maxB = fdec(keys[3]);
    float sA = (maxA - minA) / 255.0f;
    float zA = rintf(-minA / sA);
    float sB = (maxB - minB) / 255.0f;
    float zB = rintf(-minB / sB);
    scal[0] = sA; scal[1] = zA; scal[2] = sB; scal[3] = zB;
}

__global__ __launch_bounds__(256) void quantA_kernel(const float* __restrict__ x,
                                                     unsigned* __restrict__ q,
                                                     const float* __restrict__ scal, int n4) {
    const float s = scal[0], z = scal[1];
    const float4* v = (const float4*)x;
    for (int i = blockIdx.x * 256 + threadIdx.x; i < n4; i += gridDim.x * 256) {
        float4 f = v[i];
        unsigned b0 = (unsigned)(int)fminf(fmaxf(rintf(f.x / s) + z, 0.0f), 255.0f);
        unsigned b1 = (unsigned)(int)fminf(fmaxf(rintf(f.y / s) + z, 0.0f), 255.0f);
        unsigned b2 = (unsigned)(int)fminf(fmaxf(rintf(f.z / s) + z, 0.0f), 255.0f);
        unsigned b3 = (unsigned)(int)fminf(fmaxf(rintf(f.w / s) + z, 0.0f), 255.0f);
        q[i] = b0 | (b1 << 8) | (b2 << 16) | (b3 << 24);
    }
}

__global__ __launch_bounds__(256) void quantBT_kernel(const float* __restrict__ B,
                                                      unsigned char* __restrict__ qBT,
                                                      const float* __restrict__ scal,
                                                      int K, int N) {
    __shared__ unsigned char tile[64 * 68];
    const float s = scal[2], z = scal[3];
    const int t = threadIdx.x;
    const int n0 = blockIdx.x * 64;
    const int k0 = blockIdx.y * 64;
#pragma unroll
    for (int it = 0; it < 16; ++it) {
        int l = it * 256 + t;
        int r = l >> 6;
        int c = l & 63;
        float f = B[(size_t)(k0 + r) * N + n0 + c];
        float qv = fminf(fmaxf(rintf(f / s) + z, 0.0f), 255.0f);
        tile[r * 68 + c] = (unsigned char)(int)qv;
    }
    __syncthreads();
#pragma unroll
    for (int it = 0; it < 4; ++it) {
        int l = it * 256 + t;
        int r = l >> 4;
        int c = (l & 15) * 4;
        unsigned b0 = tile[(c + 0) * 68 + r];
        unsigned b1 = tile[(c + 1) * 68 + r];
        unsigned b2 = tile[(c + 2) * 68 + r];
        unsigned b3 = tile[(c + 3) * 68 + r];
        *(unsigned*)(qBT + (size_t)(n0 + r) * K + k0 + c) =
            b0 | (b1 << 8) | (b2 << 16) | (b3 << 24);
    }
}

__global__ __launch_bounds__(256) void rowsum_kernel(const unsigned char* __restrict__ src,
                                                     float* __restrict__ out, int rowBytes) {
    __shared__ unsigned sred[256];
    const int row = blockIdx.x;
    const int t = threadIdx.x;
    const uint2* p = (const uint2*)(src + (size_t)row * rowBytes);
    uint2 u = p[t];
    unsigned s = (u.x & 255u) + ((u.x >> 8) & 255u) + ((u.x >> 16) & 255u) + (u.x >> 24)
               + (u.y & 255u) + ((u.y >> 8) & 255u) + ((u.y >> 16) & 255u) + (u.y >> 24);
    sred[t] = s;
    __syncthreads();
    for (int off = 128; off > 0; off >>= 1) {
        if (t < off) sred[t] += sred[t + off];
        __syncthreads();
    }
    if (t == 0) out[row] = (float)sred[0];
}

#define TILE_M 128
#define TILE_N 128
#define TILE_K 64
#define LDSS   80

__global__ __launch_bounds__(256) void gemm_iu8_kernel(
    const unsigned char* __restrict__ qA,
    const unsigned char* __restrict__ qBT,
    const float* __restrict__ rowsumA,
    const float* __restrict__ colsumB,
    const float* __restrict__ scal,
    float* __restrict__ out, int M, int N, int K) {
    __shared__ unsigned char sA[TILE_M * LDSS];
    __shared__ unsigned char sB[TILE_N * LDSS];

    const int t = threadIdx.x;
    const int lane = t & 31;
    const int wid = t >> 5;
    const int waveM = wid & 3;
    const int waveN = wid >> 2;
    const int m0 = blockIdx.y * TILE_M;
    const int n0 = blockIdx.x * TILE_N;

    v8i acc[2][4];
#pragma unroll
    for (int mi = 0; mi < 2; ++mi)
#pragma unroll
        for (int ni = 0; ni < 4; ++ni) acc[mi][ni] = v8i{};

    const int ldRow = t >> 1;
    const int ldOff = (t & 1) * 32;

    const int laneM = lane & 15;
    const int aK = (lane >> 4) << 3;
    const int bK = (lane >> 4) << 3;

    for (int k0 = 0; k0 < K; k0 += TILE_K) {
        const uint4* gA = (const uint4*)(qA + (size_t)(m0 + ldRow) * K + k0 + ldOff);
        const uint4* gB = (const uint4*)(qBT + (size_t)(n0 + ldRow) * K + k0 + ldOff);
        uint4 a0 = gA[0], a1 = gA[1];
        uint4 b0 = gB[0], b1 = gB[1];
        *(uint4*)(sA + ldRow * LDSS + ldOff) = a0;
        *(uint4*)(sA + ldRow * LDSS + ldOff + 16) = a1;
        *(uint4*)(sB + ldRow * LDSS + ldOff) = b0;
        *(uint4*)(sB + ldRow * LDSS + ldOff + 16) = b1;
        __syncthreads();

        v8i afrag[2];
#pragma unroll
        for (int mi = 0; mi < 2; ++mi) {
            const unsigned char* base = sA + (waveM * 32 + mi * 16 + laneM) * LDSS + aK;
            uint2 c0 = *(const uint2*)(base);
            uint2 c1 = *(const uint2*)(base + 16);
            uint2 c2 = *(const uint2*)(base + 32);
            uint2 c3 = *(const uint2*)(base + 48);
            v8i a;
            a[0] = (int)c0.x; a[1] = (int)c0.y;
            a[2] = (int)c1.x; a[3] = (int)c1.y;
            a[4] = (int)c2.x; a[5] = (int)c2.y;
            a[6] = (int)c3.x; a[7] = (int)c3.y;
            afrag[mi] = a;
        }
        v8i bfrag[4];
#pragma unroll
        for (int ni = 0; ni < 4; ++ni) {
            const unsigned char* base = sB + (waveN * 64 + ni * 16 + laneM) * LDSS + bK;
            uint2 c0 = *(const uint2*)(base);
            uint2 c1 = *(const uint2*)(base + 16);
            uint2 c2 = *(const uint2*)(base + 32);
            uint2 c3 = *(const uint2*)(base + 48);
            v8i b;
            b[0] = (int)c0.x; b[1] = (int)c0.y; b[2] = (int)c1.x; b[3] = (int)c1.y;
            b[4] = (int)c2.x; b[5] = (int)c2.y; b[6] = (int)c3.x; b[7] = (int)c3.y;
            bfrag[ni] = b;
        }
#pragma unroll
        for (int mi = 0; mi < 2; ++mi)
#pragma unroll
            for (int ni = 0; ni < 4; ++ni)
                acc[mi][ni] = __builtin_amdgcn_wmma_i32_16x16x64_iu8(
                     false, afrag[mi],  false, bfrag[ni],
                    acc[mi][ni],  false,  false);
        __syncthreads();
    }

    const float sa = scal[0], za = scal[1], sb = scal[2], zb = scal[3];
    const float sab = sa * sb;
    const float corr = (float)K * za * zb;
#pragma unroll
    for (int mi = 0; mi < 2; ++mi) {
        const int rbase = m0 + waveM * 32 + mi * 16 + ((lane >> 4) << 3);
        float rs[8];
#pragma unroll
        for (int g = 0; g < 8; ++g) rs[g] = rowsumA[rbase + g];
#pragma unroll
        for (int ni = 0; ni < 4; ++ni) {
            const int col = n0 + waveN * 64 + ni * 16 + laneM;
            const float cs = colsumB[col];
#pragma unroll
            for (int g = 0; g < 8; ++g) {
                float v = sab * ((float)acc[mi][ni][g] - za * cs - zb * rs[g] + corr);
                out[(size_t)(rbase + g) * N + col] = v;
            }
        }
    }
}

extern "C" void kernel_launch(void* const* d_in, const int* in_sizes, int n_in,
                              void* d_out, int out_size, void* d_ws, size_t ws_size,
                              hipStream_t stream) {
    const float* A = (const float*)d_in[0];
    const float* B = (const float*)d_in[1];
    float* out = (float*)d_out;
    const int M = 4096, N = 4096, K = 2048;

    char* ws = (char*)d_ws;
    unsigned* keys = (unsigned*)ws;
    float* scal = (float*)(ws + 64);
    float* rowsumA = (float*)(ws + 256);
    float* colsumB = (float*)(ws + 256 + 4096 * 4);
    unsigned char* qA = (unsigned char*)(ws + 65536);
    unsigned char* qBT = qA + (size_t)M * K;

    init_keys_kernel<<<1, 1, 0, stream>>>(keys);
    minmax_kernel<<<1024, 256, 0, stream>>>(A, M * K / 4, keys);
    minmax_kernel<<<1024, 256, 0, stream>>>(B, K * N / 4, keys + 2);
    finalize_kernel<<<1, 1, 0, stream>>>(keys, scal);

    quantA_kernel<<<2048, 256, 0, stream>>>(A, (unsigned*)qA, scal, M * K / 4);
    quantBT_kernel<<<dim3(N / 64, K / 64), 256, 0, stream>>>(B, qBT, scal, K, N);

    rowsum_kernel<<<M, 256, 0, stream>>>(qA, rowsumA, K);
    rowsum_kernel<<<N, 256, 0, stream>>>(qBT, colsumB, K);

    gemm_iu8_kernel<<<dim3(N / TILE_N, M / TILE_M), 256, 0, stream>>>(
        qA, qBT, rowsumA, colsumB, scal, out, M, N, K);
}
